// HeteroGCN_57432302682299
// MI455X (gfx1250) — matrix operand order repaired and checked by reading, not compiled, not run on hardware
//
#include <hip/hip_runtime.h>
#include <hip/hip_bf16.h>

typedef __attribute__((ext_vector_type(16))) __bf16 v16bf;
typedef __attribute__((ext_vector_type(8)))  __bf16 v8bf;
typedef __attribute__((ext_vector_type(2)))  __bf16 v2bf;
typedef __attribute__((ext_vector_type(8)))  float  v8f;
typedef __attribute__((ext_vector_type(4)))  int    v4i;

#define D_FEAT 128
#define H_FEAT 128

// --- optional CDNA5 async global->LDS path (guarded; falls back to sync copy)
#if defined(__has_builtin)
#if __has_builtin(__builtin_amdgcn_global_load_async_to_lds_b128) && \
    __has_builtin(__builtin_amdgcn_s_wait_asynccnt)
#define HAVE_ASYNC_LDS 1
#endif
#endif
#ifndef HAVE_ASYNC_LDS
#define HAVE_ASYNC_LDS 0
#endif

#if HAVE_ASYNC_LDS
// clang prints AS1 as __device__, AS3 as __shared__; builtin wants v4i pointers.
typedef __attribute__((address_space(1))) v4i *g_v4i_p;
typedef __attribute__((address_space(3))) v4i *l_v4i_p;
#endif

__device__ __forceinline__ void copy16_to_lds(const void* g, void* l) {
#if HAVE_ASYNC_LDS
    __builtin_amdgcn_global_load_async_to_lds_b128(
        (g_v4i_p)(unsigned long long)g,
        (l_v4i_p)(unsigned int)(unsigned long long)l, 0, 0);
#else
    *(uint4*)l = *(const uint4*)g;
#endif
}

__device__ __forceinline__ void wait_lds_copies() {
#if HAVE_ASYNC_LDS
    __builtin_amdgcn_s_wait_asynccnt(0);
#endif
}

// ---------------------------------------------------------------------------
// Zero-init scratch (s accumulator + degree counts).
// ---------------------------------------------------------------------------
__global__ __launch_bounds__(256) void sage_zero_kernel(float* __restrict__ p, long n) {
    long i = (long)blockIdx.x * blockDim.x + threadIdx.x;
    long stride = (long)gridDim.x * blockDim.x;
    for (; i < n; i += stride) p[i] = 0.0f;
}

// ---------------------------------------------------------------------------
// x (f32) -> bf16 mirror, once per launch. Each thread converts 2 floats.
// ---------------------------------------------------------------------------
__global__ __launch_bounds__(256) void sage_xconv_kernel(
    const float* __restrict__ x, __bf16* __restrict__ xb, long n2) {
    long i = (long)blockIdx.x * blockDim.x + threadIdx.x;
    if (i >= n2) return;
    float2 v = ((const float2*)x)[i];
    v2bf o; o[0] = (__bf16)v.x; o[1] = (__bf16)v.y;
    ((v2bf*)xb)[i] = o;
}

// ---------------------------------------------------------------------------
// h_neigh = s / max(deg,1) -> bf16, folding normalization (v_rcp_f32, no div).
// ---------------------------------------------------------------------------
__global__ __launch_bounds__(256) void sage_norm_kernel(
    const float* __restrict__ s, const float* __restrict__ deg,
    __bf16* __restrict__ hb, long n2) {
    long i = (long)blockIdx.x * blockDim.x + threadIdx.x;
    if (i >= n2) return;
    const int row = (int)(i >> 6);                 // 64 float2 per 128-wide row
    const float inv = __builtin_amdgcn_rcpf(fmaxf(deg[row], 1.0f));
    float2 v = ((const float2*)s)[i];
    v2bf o; o[0] = (__bf16)(v.x * inv); o[1] = (__bf16)(v.y * inv);
    ((v2bf*)hb)[i] = o;
}

// ---------------------------------------------------------------------------
// Weights: transpose + convert to bf16 [n][k] so a WMMA B-fragment is one
// contiguous 32B load. 128x128 each, both matrices in one tiny kernel.
// ---------------------------------------------------------------------------
__global__ __launch_bounds__(256) void sage_wconv_kernel(
    const float* __restrict__ Ws, const float* __restrict__ Wn,
    __bf16* __restrict__ wts, __bf16* __restrict__ wtn) {
    const int t = blockIdx.x * 256 + threadIdx.x;          // 16384 threads
    const int k = t >> 7, n = t & 127;
    wts[n * D_FEAT + k] = (__bf16)Ws[t];
    wtn[n * D_FEAT + k] = (__bf16)Wn[t];
}

// ---------------------------------------------------------------------------
// Edge scatter: one wave32 per edge; lane -> float4 of the feature row.
// Coalesced 512B read of x[src], 4 global_atomic_add_f32 into s[dst].
// x and s both live in the 192MB L2, so atomics resolve at L2 rate.
// ---------------------------------------------------------------------------
__global__ __launch_bounds__(256) void sage_scatter_kernel(
    const float* __restrict__ x, const int* __restrict__ src,
    const int* __restrict__ dst, const float* __restrict__ w,
    float* __restrict__ s, float* __restrict__ deg, int E)
{
    const int lane = threadIdx.x & 31;
    const int e = (int)(((long)blockIdx.x * blockDim.x + threadIdx.x) >> 5);
    if (e >= E) return;
    const int   u  = src[e];
    const int   v  = dst[e];
    const float wt = w[e];
    const float4 xv = *(const float4*)(x + (size_t)u * D_FEAT + lane * 4);
    float* sp = s + (size_t)v * D_FEAT + lane * 4;
    atomicAdd(sp + 0, xv.x * wt);
    atomicAdd(sp + 1, xv.y * wt);
    atomicAdd(sp + 2, xv.z * wt);
    atomicAdd(sp + 3, xv.w * wt);
    if (lane == 0) atomicAdd(deg + v, 1.0f);
}

// ---------------------------------------------------------------------------
// Fused dual-GEMM + bias + ReLU + (1/3) accumulate via v_wmma_f32_16x16x32_bf16.
// Block = one 16-row stripe; 8 waves = 8 column tiles of 16. The 16x128 bf16
// x / h_neigh tiles are staged once into LDS (async global->LDS when
// available) since all 8 waves share them; B fragments are contiguous 32B
// loads from the pre-transposed bf16 weights.
// ---------------------------------------------------------------------------
__global__ __launch_bounds__(256) void sage_wmma_gemm_kernel(
    const __bf16* __restrict__ xb, const __bf16* __restrict__ hb,
    const __bf16* __restrict__ wts, const __bf16* __restrict__ wtn,
    const float* __restrict__ bias, float* __restrict__ out, int first)
{
    __shared__ __align__(16) __bf16 xs[16 * D_FEAT];   // 4KB
    __shared__ __align__(16) __bf16 hs[16 * D_FEAT];   // 4KB

    const int tid  = threadIdx.x;
    const int m0   = blockIdx.x * 16;                  // N = 6250 * 16 exactly

    // Stage both tiles: 256 threads x 16B each per tile.
    {
        const int row  = tid >> 4;
        const int col8 = (tid & 15) * 8;
        const size_t goff = (size_t)(m0 + row) * D_FEAT + col8;
        const int    loff = row * D_FEAT + col8;
        copy16_to_lds(xb + goff, xs + loff);
        copy16_to_lds(hb + goff, hs + loff);
        wait_lds_copies();
    }
    __syncthreads();

    const int wave = tid >> 5;                         // col tile
    const int lane = tid & 31;
    const int h0   = wave * 16;
    const int lr   = lane & 15;
    const bool hi  = lane >= 16;
    const int Ncol = h0 + lr;
    const float bN = bias[Ncol];

    const int khalfA = hi ? 16 : 0;                    // A: contiguous k = 16h + s
    const int khalfB = hi ? 16 : 0;                    // B: 16-bit 32x16 layout

    v8f c = {};
#pragma unroll
    for (int kk = 0; kk < D_FEAT; kk += 32) {
        const int abase = lr * D_FEAT + kk + khalfA;
        v8bf xlo = *(const v8bf*)(xs + abase);
        v8bf xhi = *(const v8bf*)(xs + abase + 8);
        v8bf hlo = *(const v8bf*)(hs + abase);
        v8bf hhi = *(const v8bf*)(hs + abase + 8);
        v16bf xa = __builtin_shufflevector(xlo, xhi, 0,1,2,3,4,5,6,7,
                                                     8,9,10,11,12,13,14,15);
        v16bf ha = __builtin_shufflevector(hlo, hhi, 0,1,2,3,4,5,6,7,
                                                     8,9,10,11,12,13,14,15);
        const size_t bbase = (size_t)Ncol * D_FEAT + kk + khalfB;
        v16bf wsb = *(const v16bf*)(wts + bbase);
        v16bf wnb = *(const v16bf*)(wtn + bbase);
        c = __builtin_amdgcn_wmma_f32_16x16x32_bf16(
                false, xa, false, wsb, (short)0, c, false, false);
        c = __builtin_amdgcn_wmma_f32_16x16x32_bf16(
                false, ha, false, wnb, (short)0, c, false, false);
    }

#pragma unroll
    for (int v = 0; v < 8; ++v) {
        const int Mo = m0 + v + (hi ? 8 : 0);
        float val = fmaxf(c[v] + bN, 0.0f) * (1.0f / 3.0f);
        const size_t idx = (size_t)Mo * H_FEAT + Ncol;
        if (first) out[idx] = val;
        else       out[idx] += val;
    }
}

// ---------------------------------------------------------------------------
// Launch. Input dict order: x, then per r: src, dst, w, Wself, Wneigh, b.
// ws layout: s (N*128 f32) | deg (N f32) | xb | hb (N*128 bf16 each) |
//            wts | wtn (128*128 bf16 each)  -> ~103 MB total.
// ---------------------------------------------------------------------------
extern "C" void kernel_launch(void* const* d_in, const int* in_sizes, int n_in,
                              void* d_out, int out_size, void* d_ws, size_t ws_size,
                              hipStream_t stream) {
    const float* x = (const float*)d_in[0];
    const int N = in_sizes[0] / D_FEAT;
    float* out = (float*)d_out;

    float*  s   = (float*)d_ws;
    float*  deg = s + (size_t)N * D_FEAT;
    __bf16* xb  = (__bf16*)(deg + N);
    __bf16* hb  = xb + (size_t)N * D_FEAT;
    __bf16* wts = hb + (size_t)N * D_FEAT;
    __bf16* wtn = wts + D_FEAT * H_FEAT;

    const long n2 = (long)N * (D_FEAT / 2);            // float2 elements
    const int  cblocks = (int)((n2 + 255) / 256);

    sage_xconv_kernel<<<cblocks, 256, 0, stream>>>(x, xb, n2);

    for (int r = 0; r < 3; ++r) {
        const int*   src    = (const int*)  d_in[1 + 6 * r + 0];
        const int*   dst    = (const int*)  d_in[1 + 6 * r + 1];
        const float* w      = (const float*)d_in[1 + 6 * r + 2];
        const float* Wself  = (const float*)d_in[1 + 6 * r + 3];
        const float* Wneigh = (const float*)d_in[1 + 6 * r + 4];
        const float* bias   = (const float*)d_in[1 + 6 * r + 5];
        const int E = in_sizes[1 + 6 * r + 0];

        sage_zero_kernel<<<2048, 256, 0, stream>>>(s, (long)N * D_FEAT + N);

        sage_scatter_kernel<<<(E + 7) / 8, 256, 0, stream>>>(
            x, src, dst, w, s, deg, E);

        sage_norm_kernel<<<cblocks, 256, 0, stream>>>(s, deg, hb, n2);

        sage_wconv_kernel<<<(D_FEAT * H_FEAT) / 256, 256, 0, stream>>>(
            Wself, Wneigh, wts, wtn);

        sage_wmma_gemm_kernel<<<N / 16, 256, 0, stream>>>(
            xb, hb, wts, wtn, bias, out, r == 0 ? 1 : 0);
    }
}
